// RNN_83846351552987
// MI455X (gfx1250) — matrix operand order repaired and checked by reading, not compiled, not run on hardware
//
#include <hip/hip_runtime.h>
#include <hip/hip_bf16.h>
#include <math.h>

// RNN: T=2048, B=64, I=H=O=512.  bf16 WMMA (16x16x32) with f32 accumulation.
#define T_STEPS 2048
#define BATCH   64
#define HID     512

typedef __attribute__((ext_vector_type(16))) __bf16 v16bf;
typedef __attribute__((ext_vector_type(8)))  float  v8f;
typedef __attribute__((ext_vector_type(4)))  int    v4i_;

#define AS_GLOBAL __attribute__((address_space(1)))
#define AS_LDS    __attribute__((address_space(3)))

// ---- CDNA5 async global->LDS copy (ASYNCcnt path), with safe fallback ----
#if __has_builtin(__builtin_amdgcn_global_load_async_to_lds_b128) && \
    __has_builtin(__builtin_amdgcn_s_wait_asynccnt)
#define USE_ASYNC_LDS 1
#else
#define USE_ASYNC_LDS 0
#endif

__device__ __forceinline__ void lds_copy_b128(void* l, const void* g) {
#if USE_ASYNC_LDS
  // signature (from hipcc diagnostics): (int4 AS1*, int4 AS3*, imm off, imm cpol)
  __builtin_amdgcn_global_load_async_to_lds_b128(
      (AS_GLOBAL v4i_*)g, (AS_LDS v4i_*)l, /*imm offset*/ 0, /*cpol*/ 0);
#else
  *(uint4*)l = *(const uint4*)g;
#endif
}

__device__ __forceinline__ void lds_copy_fence() {
#if USE_ASYNC_LDS
  __builtin_amdgcn_s_wait_asynccnt(0);
#endif
}

__device__ __forceinline__ v8f zero_v8f() {
  v8f z;
#pragma unroll
  for (int i = 0; i < 8; ++i) z[i] = 0.0f;
  return z;
}

__device__ __forceinline__ v8f wmma_bf16(v16bf a, v16bf b, v8f c) {
  // (neg_a, A, neg_b, B, c_mod, C, reuse_a, reuse_b)
  return __builtin_amdgcn_wmma_f32_16x16x32_bf16(false, a, false, b, (short)0, c,
                                                 false, false);
}

// Single-instruction tanh on CDNA5 (V_TANH trans op); branchless fallback.
__device__ __forceinline__ float fast_tanh(float x) {
#if __has_builtin(__builtin_amdgcn_tanhf)
  return __builtin_amdgcn_tanhf(x);
#elif __has_builtin(__builtin_amdgcn_tanh_f32)
  return __builtin_amdgcn_tanh_f32(x);
#else
  const float ax = fabsf(x);
  const float e  = __expf(2.0f * ax);         // v_exp_f32-based, no branches
  const float r  = 1.0f - 2.0f / (e + 1.0f);  // e==inf -> r==1
  return copysignf(r, x);
#endif
}

// ---- WMMA fragment loaders (layouts per CDNA5 ISA 7.12.2) ----
// A: 16x32 bf16, source row-major [*, lda]. Per lane: row m0+(lane&15),
// halves k..k+7 and k+16..k+23 (lanes>=16 shifted by 8 in K).
__device__ __forceinline__ v16bf load_a_frag_bf16(const __bf16* A, int lda,
                                                  int m0, int k0, int lane) {
  const int m  = m0 + (lane & 15);
  const int kb = k0 + ((lane >> 4) << 3);
  const __bf16* p = A + (size_t)m * lda + kb;
  union { uint4 u[2]; v16bf v; } c;
  c.u[0] = *(const uint4*)(p);
  c.u[1] = *(const uint4*)(p + 16);
  return c.v;
}

// B: 32x16 with B[k][n] = W[n][k], W row-major [N][K]. Per lane: row
// n0+(lane&15), halves k..k+7 and k+16..k+23 (lanes>=16 shifted by 8 in K).
__device__ __forceinline__ v16bf load_b_frag(const __bf16* W, int ldw,
                                             int n0, int k0, int lane) {
  const int n = n0 + (lane & 15);
  const int k = k0 + ((lane >> 4) << 3);
  const __bf16* p = W + (size_t)n * ldw + k;
  union { uint4 u[2]; v16bf v; } c;
  c.u[0] = *(const uint4*)(p);
  c.u[1] = *(const uint4*)(p + 16);
  return c.v;
}

// ---- one-shot fp32 -> bf16 conversion of the three weight matrices + h0 ----
__global__ void cvt_weights_kernel(const float* __restrict__ W_ih,
                                   const float* __restrict__ W_hh,
                                   const float* __restrict__ W_out,
                                   const float* __restrict__ h0,
                                   __bf16* __restrict__ Wih_b,
                                   __bf16* __restrict__ Whh_b,
                                   __bf16* __restrict__ Wout_b,
                                   __bf16* __restrict__ hseq0) {
  const int i = blockIdx.x * blockDim.x + threadIdx.x;
  if (i < HID * HID) {
    Wih_b[i]  = (__bf16)W_ih[i];
    Whh_b[i]  = (__bf16)W_hh[i];
    Wout_b[i] = (__bf16)W_out[i];
  }
  if (i < BATCH * HID) hseq0[i] = (__bf16)h0[i];
}

// ---- streaming fp32 -> bf16 conversion of x (memory bound, ~0.4 GB) ----
__global__ __launch_bounds__(256)
void cvt_x_kernel(const float* __restrict__ xin, __bf16* __restrict__ xb,
                  long n4) {
  long i = blockIdx.x * (long)blockDim.x + threadIdx.x;
  const long stride = (long)gridDim.x * blockDim.x;
  for (; i < n4; i += stride) {
    const float4 f = ((const float4*)xin)[i];
    __bf16* o = xb + 4 * i;
    o[0] = (__bf16)f.x;
    o[1] = (__bf16)f.y;
    o[2] = (__bf16)f.z;
    o[3] = (__bf16)f.w;
  }
}

// ---- big parallel GEMM: out[M x 512] = A[M x 512] @ W^T (+ bias) ----
// 256 threads = 8 waves; each wave owns 16 rows, loops over 8 chunks of 64
// cols. Each 64-row chunk of W is staged once into LDS via async DMA and
// shared by all 8 waves (8x cut of L2 fragment traffic).
__global__ __launch_bounds__(256)
void gemm512_kernel(const __bf16* __restrict__ A, const __bf16* __restrict__ W,
                    const float* __restrict__ bias, float* __restrict__ out) {
  __shared__ __align__(32) __bf16 Bs[64 * HID];  // 64 KB W chunk
  const int tid  = threadIdx.x;
  const int lane = tid & 31;
  const int wave = tid >> 5;
  const int m0 = blockIdx.x * 128 + wave * 16;
#pragma unroll 1
  for (int nc = 0; nc < 8; ++nc) {
    const int n0 = nc * 64;
    __syncthreads();  // all readers of the previous chunk are done
    {
      const __bf16* src = W + (size_t)n0 * HID;
#pragma unroll
      for (int i = tid; i < (64 * HID) / 8; i += 256)
        lds_copy_b128(Bs + 8 * i, src + 8 * i);
      lds_copy_fence();  // s_wait_asynccnt 0
    }
    __syncthreads();

    v8f acc[4];
#pragma unroll
    for (int j = 0; j < 4; ++j) acc[j] = zero_v8f();
#pragma unroll 2
    for (int k0 = 0; k0 < HID; k0 += 32) {
      const v16bf a = load_a_frag_bf16(A, HID, m0, k0, lane);
#pragma unroll
      for (int j = 0; j < 4; ++j) {
        const v16bf b = load_b_frag(Bs, HID, 16 * j, k0, lane);
        acc[j] = wmma_bf16(a, b, acc[j]);
      }
    }
    // C/D layout: VGPR r of lane L -> (M = ((L>>4)<<3)+r, N = L&15)
    const int mg = m0 + ((lane >> 4) << 3);
    const int nl = lane & 15;
#pragma unroll
    for (int j = 0; j < 4; ++j) {
      const int col = n0 + 16 * j + nl;
      const float bv = bias ? bias[col] : 0.0f;
#pragma unroll
      for (int r = 0; r < 8; ++r) {
        out[(size_t)(mg + r) * HID + col] = acc[j][r] + bv;
      }
    }
  }
}

// ---- sequential recurrence: 16 persistent WGs, one 32-col slice of h each ----
// h_seq[t+1] = tanh(xp[t] + h_seq[t] @ W_hh^T + b_ih + b_hh), shared via L2.
__global__ __launch_bounds__(128)
void rnn_recurrence_kernel(const float* __restrict__ xp,
                           const __bf16* __restrict__ Whh,
                           const float* __restrict__ b_ih,
                           const float* __restrict__ b_hh,
                           __bf16* __restrict__ h_seq,
                           unsigned* __restrict__ bar) {
  __shared__ __align__(32) __bf16 Ws[32 * HID];  // 32 KB slice of W_hh in LDS
  const int tid  = threadIdx.x;
  const int lane = tid & 31;
  const int wave = tid >> 5;
  const int n0   = blockIdx.x * 32;  // this WG's output columns of h
  const unsigned nwg = gridDim.x;

  // Stage W_hh rows [n0, n0+32) into LDS once (reused 2048 times).
  {
    const __bf16* src = Whh + (size_t)n0 * HID;
#pragma unroll
    for (int i = tid; i < (32 * HID) / 8; i += 128)
      lds_copy_b128(Ws + 8 * i, src + 8 * i);
    lds_copy_fence();
  }
  __syncthreads();

  const int mLoc = wave * 16;                 // 4 waves cover B=64 rows
  const int mg   = mLoc + ((lane >> 4) << 3);
  const int nl   = lane & 15;
  const int c0   = n0 + nl;
  const int c1   = n0 + 16 + nl;
  const float bc0 = b_ih[c0] + b_hh[c0];
  const float bc1 = b_ih[c1] + b_hh[c1];

  for (int t = 0; t < T_STEPS; ++t) {
    const __bf16* hprev = h_seq + (size_t)t * BATCH * HID;
    v8f acc0 = zero_v8f();
    v8f acc1 = zero_v8f();
#pragma unroll 4
    for (int k0 = 0; k0 < HID; k0 += 32) {
      const v16bf a  = load_a_frag_bf16(hprev, HID, mLoc, k0, lane);
      const v16bf b0 = load_b_frag(Ws, HID, 0,  k0, lane);
      const v16bf b1 = load_b_frag(Ws, HID, 16, k0, lane);
      acc0 = wmma_bf16(a, b0, acc0);
      acc1 = wmma_bf16(a, b1, acc1);
    }
    const float* xpt = xp + (size_t)t * BATCH * HID;
    __bf16* hout = h_seq + (size_t)(t + 1) * BATCH * HID;
#pragma unroll
    for (int r = 0; r < 8; ++r) {
      const int m = mg + r;
      const float v0 = fast_tanh(acc0[r] + xpt[(size_t)m * HID + c0] + bc0);
      const float v1 = fast_tanh(acc1[r] + xpt[(size_t)m * HID + c1] + bc1);
      hout[(size_t)m * HID + c0] = (__bf16)v0;
      hout[(size_t)m * HID + c1] = (__bf16)v1;
    }
    if (t + 1 < T_STEPS)  // warm L2/WGP$ for next step's xp tile
      __builtin_prefetch(xp + (size_t)(t + 1) * BATCH * HID + (size_t)mg * HID + c0, 0, 0);

    // Device-wide step barrier. The cluster barrier is a hardware fast path
    // when dispatched as a 16-WG cluster and an architectural NOP otherwise
    // (ISA 3.2: ClusterID==0 -> barrier ops become NOP).
    __syncthreads();
    __builtin_amdgcn_s_cluster_barrier();
    if (tid == 0) {
      __hip_atomic_fetch_add(bar, 1u, __ATOMIC_RELEASE, __HIP_MEMORY_SCOPE_AGENT);
      const unsigned target = (unsigned)(t + 1) * nwg;  // monotonic, no reset race
      while (__hip_atomic_load(bar, __ATOMIC_ACQUIRE, __HIP_MEMORY_SCOPE_AGENT) < target) {
        __builtin_amdgcn_s_sleep(2);
      }
    }
    __syncthreads();
  }
}

extern "C" void kernel_launch(void* const* d_in, const int* in_sizes, int n_in,
                              void* d_out, int out_size, void* d_ws, size_t ws_size,
                              hipStream_t stream) {
  const float* x     = (const float*)d_in[0];
  const float* h0    = (const float*)d_in[1];
  const float* W_ih  = (const float*)d_in[2];
  const float* b_ih  = (const float*)d_in[3];
  const float* W_hh  = (const float*)d_in[4];
  const float* b_hh  = (const float*)d_in[5];
  const float* W_out = (const float*)d_in[6];
  const float* b_out = (const float*)d_in[7];
  float* y = (float*)d_out;

  // Workspace layout (~538 MB):
  char* ws = (char*)d_ws;
  size_t off = 0;
  __bf16* Whh_b  = (__bf16*)(ws + off); off += (size_t)HID * HID * 2;      // 512 KB
  __bf16* Wih_b  = (__bf16*)(ws + off); off += (size_t)HID * HID * 2;      // 512 KB
  __bf16* Wout_b = (__bf16*)(ws + off); off += (size_t)HID * HID * 2;      // 512 KB
  unsigned* bar  = (unsigned*)(ws + off); off += 256;                      // barrier
  __bf16* h_seq  = (__bf16*)(ws + off);
  off += (size_t)(T_STEPS + 1) * BATCH * HID * 2;                          // 134 MB
  float* xp      = (float*)(ws + off);
  off += (size_t)T_STEPS * BATCH * HID * 4;                                // 268 MB
  __bf16* x_b    = (__bf16*)(ws + off);
  off += (size_t)T_STEPS * BATCH * HID * 2;                                // 134 MB
  (void)off; (void)ws_size; (void)in_sizes; (void)n_in; (void)out_size;

  (void)hipMemsetAsync(bar, 0, 256, stream);  // barrier counter starts at 0 each call

  cvt_weights_kernel<<<(HID * HID + 255) / 256, 256, 0, stream>>>(
      W_ih, W_hh, W_out, h0, Wih_b, Whh_b, Wout_b, h_seq);

  // x -> bf16 once (read exactly once from HBM in fp32)
  cvt_x_kernel<<<4096, 256, 0, stream>>>(x, x_b,
                                         (long)T_STEPS * BATCH * HID / 4);

  // Phase 1: xp = x @ W_ih^T   (bias folded into recurrence epilogue)
  gemm512_kernel<<<(T_STEPS * BATCH) / 128, 256, 0, stream>>>(
      x_b, Wih_b, nullptr, xp);

  // Phase 2: sequential recurrence across 2048 steps, 16 persistent WGs
  rnn_recurrence_kernel<<<HID / 32, 128, 0, stream>>>(xp, Whh_b, b_ih, b_hh,
                                                      h_seq, bar);

  // Phase 3: y = h_all @ W_out^T + b_out
  gemm512_kernel<<<(T_STEPS * BATCH) / 128, 256, 0, stream>>>(
      h_seq + BATCH * HID, Wout_b, b_out, y);
}
